// NodeLayer_33852932227353
// MI455X (gfx1250) — matrix operand order repaired and checked by reading, not compiled, not run on hardware
//
#include <hip/hip_runtime.h>
#include <hip/hip_bf16.h>

#define N_NODES 50000
#define N_EDGES 800000

typedef __attribute__((ext_vector_type(16))) __bf16 v16bf;
typedef __attribute__((ext_vector_type(8)))  __bf16 v8bf;
typedef __attribute__((ext_vector_type(4)))  __bf16 v4bf;
typedef __attribute__((ext_vector_type(8)))  float  v8f;

// ---------- monotone float<->uint key for atomicMax on floats ----------
__device__ __forceinline__ unsigned fkey(float f) {
    unsigned b = __float_as_uint(f);
    return (b & 0x80000000u) ? ~b : (b | 0x80000000u);
}
__device__ __forceinline__ float funkey(unsigned k) {
    unsigned b = (k & 0x80000000u) ? (k & 0x7FFFFFFFu) : ~k;
    return __uint_as_float(b);
}

// ---------- degree / count ----------
__global__ __launch_bounds__(256) void k_deg(const int* __restrict__ ei, int E,
                                             int* __restrict__ degRow, int* __restrict__ cnt) {
    int e = blockIdx.x * 256 + threadIdx.x;
    if (e < E) {
        atomicAdd(&degRow[ei[e]], 1);      // row (source) counts -> BN0 stats
        atomicAdd(&cnt[ei[E + e]], 1);     // col (target) counts -> scatter-mean
    }
}

// ---------- BN0 stats for x-part (weighted by out-degree) + x -> bf16 ----------
__global__ __launch_bounds__(64) void k_statsx(const float* __restrict__ x,
                                               const int* __restrict__ degRow, int N,
                                               float* __restrict__ sum, float* __restrict__ sq,
                                               __bf16* __restrict__ xb) {
    int j = threadIdx.x;
    float s = 0.f, q = 0.f;
    for (int n = blockIdx.x; n < N; n += gridDim.x) {
        float v = x[(size_t)n * 64 + j];
        xb[(size_t)n * 64 + j] = (__bf16)v;
        float d = (float)degRow[n];
        s += d * v; q += d * v * v;
    }
    atomicAdd(&sum[j], s);
    atomicAdd(&sq[j], q);
}

// ---------- BN0 stats for edge_attr-part + edge_attr -> bf16 ----------
__global__ __launch_bounds__(64) void k_statse(const float* __restrict__ ea, int E,
                                               float* __restrict__ sum, float* __restrict__ sq,
                                               __bf16* __restrict__ eb) {
    int j = threadIdx.x;
    float s = 0.f, q = 0.f;
    for (int e = blockIdx.x; e < E; e += gridDim.x) {
        float v = ea[(size_t)e * 64 + j];
        eb[(size_t)e * 64 + j] = (__bf16)v;
        s += v; q += v * v;
    }
    atomicAdd(&sum[j], s);
    atomicAdd(&sq[j], q);
}

// ---------- fold BN into Linear: W'[o,j]=W*g*rsqrt(v+eps), b' absorbs shift ----------
__global__ __launch_bounds__(128) void k_fold(const float* __restrict__ sum, const float* __restrict__ sq,
                                              float invR,
                                              const float* __restrict__ g, const float* __restrict__ be,
                                              const float* __restrict__ W, const float* __restrict__ b,
                                              int K, __bf16* __restrict__ Wp, float* __restrict__ bp) {
    __shared__ float sc_[128], sh_[128];
    int t = threadIdx.x;
    if (t < K) {
        float m = sum[t] * invR;
        float v = sq[t] * invR - m * m;
        float sc = g[t] * rsqrtf(v + 1e-5f);
        sc_[t] = sc;
        sh_[t] = be[t] - sc * m;
    }
    __syncthreads();
    if (t < 64) {
        float acc = b[t];
        for (int j = 0; j < K; ++j) {
            float w = W[(size_t)t * K + j];
            acc += w * sh_[j];
            Wp[(size_t)t * K + j] = (__bf16)(w * sc_[j]);
        }
        bp[t] = acc;
    }
}

// =====================================================================
// WMMA GEMM pass: [R rows, KDIM] x W'[64, KDIM]^T + b'.  Each wave owns
// TWO 16-row tiles (32 rows) so every B fragment feeds two WMMAs; a
// 256-thread block covers 256 rows.  Optional: LReLU, bf16/f32 output
// (coalesced via LDS transpose), next-BN stats, gate (logits + segment
// atomicMax).
// =====================================================================
template <int KDIM, bool LRELU, bool OUT_BF16, bool OUT_F32, bool STATS, bool GATE>
__global__ __launch_bounds__(256) void gemm_tile(
    const __bf16* __restrict__ srcA,      // [*,64] (first 64 cols; gathered if gi)
    const __bf16* __restrict__ srcB,      // [R,64] cols 64..127 (KDIM==128 only)
    const int*    __restrict__ gi,        // gather index for srcA rows (or null)
    const __bf16* __restrict__ W,         // [64, KDIM] folded weights, bf16
    const float*  __restrict__ bias,      // [64] folded bias
    int R,
    __bf16* __restrict__ outBf,           // [R,64]
    float*  __restrict__ outF,            // [R,64]
    float* __restrict__ statsSum, float* __restrict__ statsSq,   // [64]
    const float* __restrict__ gateW, const float* __restrict__ gateB,
    float* __restrict__ logitsOut, unsigned int* __restrict__ segMax,
    const int* __restrict__ colIdx) {
    __shared__ float lds[256 * 64 + 128];      // 64KB tile + 2x64 stat partials
    float* sSum = lds + 16384;
    float* sSq  = lds + 16384 + 64;

    if (STATS && threadIdx.x < 128) lds[16384 + threadIdx.x] = 0.f;
    __syncthreads();

    const int wave = threadIdx.x >> 5;
    const int lane = threadIdx.x & 31;
    const int lm = lane & 15;          // row within tile (A) / out column (B,D)
    const int hb = lane >> 4;          // half-select per ISA fragment layout
    const int rowBase = blockIdx.x * 256 + wave * 32;

    // per-tile clamped A-row offsets
    size_t rA[2], rB[2];
#pragma unroll
    for (int t = 0; t < 2; ++t) {
        int myRow = rowBase + 16 * t + lm;
        int rClamp = myRow < R ? myRow : (R - 1);
        rA[t] = (size_t)((KDIM == 128 && gi) ? gi[rClamp] : rClamp) * 64;
        rB[t] = (size_t)rClamp * 64;
    }

    // D/C fragments: lane holds column (lm+16f), rows 8*hb .. 8*hb+7 (+16*t)
    v8f c[2][4];
#pragma unroll
    for (int f = 0; f < 4; ++f) {
        float bv = bias[lm + 16 * f];
#pragma unroll
        for (int t = 0; t < 2; ++t)
#pragma unroll
            for (int r = 0; r < 8; ++r) c[t][f][r] = bv;
    }

#pragma unroll
    for (int ks = 0; ks < KDIM / 32; ++ks) {
        // A fragments: lane (m=lm, hb) holds K = 32*ks + 8*hb + {0..7, 16..23}
        const int c0 = ks * 32 + 8 * hb;
        v16bf a[2];
#pragma unroll
        for (int t = 0; t < 2; ++t) {
            v8bf lo, hi;
            if (KDIM == 128 && ks >= 2) {          // second operand of the concat
                lo = *(const v8bf*)(srcB + rB[t] + (c0 - 64));
                hi = *(const v8bf*)(srcB + rB[t] + (c0 - 48));
            } else {
                lo = *(const v8bf*)(srcA + rA[t] + c0);
                hi = *(const v8bf*)(srcA + rA[t] + c0 + 16);
            }
            a[t] = __builtin_shufflevector(lo, hi, 0, 1, 2, 3, 4, 5, 6, 7,
                                           8, 9, 10, 11, 12, 13, 14, 15);
        }
#pragma unroll
        for (int f = 0; f < 4; ++f) {
            // B fragment: lane (n=lm, hb) holds K = 32*ks + 8*hb + {0..7, 16..23} of row o=lm+16f
            const __bf16* wp = W + (size_t)(lm + 16 * f) * KDIM + ks * 32 + hb * 8;
            v8bf blo = *(const v8bf*)(wp);
            v8bf bhi = *(const v8bf*)(wp + 16);
            v16bf b = __builtin_shufflevector(blo, bhi, 0, 1, 2, 3, 4, 5, 6, 7,
                                              8, 9, 10, 11, 12, 13, 14, 15);
            c[0][f] = __builtin_amdgcn_wmma_f32_16x16x32_bf16(
                false, a[0], false, b, (short)0, c[0][f], false, false);
            c[1][f] = __builtin_amdgcn_wmma_f32_16x16x32_bf16(
                false, a[1], false, b, (short)0, c[1][f], false, false);
        }
    }

    if (LRELU) {
#pragma unroll
        for (int t = 0; t < 2; ++t)
#pragma unroll
            for (int f = 0; f < 4; ++f)
#pragma unroll
                for (int r = 0; r < 8; ++r) {
                    float v = c[t][f][r];
                    c[t][f][r] = v > 0.f ? v : 0.1f * v;
                }
    }

    if (STATS) {
#pragma unroll
        for (int f = 0; f < 4; ++f) {
            float s = 0.f, q = 0.f;
#pragma unroll
            for (int t = 0; t < 2; ++t)
#pragma unroll
                for (int r = 0; r < 8; ++r) {
                    int grow = rowBase + 16 * t + 8 * hb + r;
                    if (grow < R) { float v = c[t][f][r]; s += v; q += v * v; }
                }
            atomicAdd(&sSum[lm + 16 * f], s);
            atomicAdd(&sSq[lm + 16 * f], q);
        }
    }

    if (GATE) {
#pragma unroll
        for (int t = 0; t < 2; ++t) {
            float p[8];
#pragma unroll
            for (int r = 0; r < 8; ++r) p[r] = 0.f;
#pragma unroll
            for (int f = 0; f < 4; ++f) {
                float gw = gateW[lm + 16 * f];
#pragma unroll
                for (int r = 0; r < 8; ++r) p[r] += c[t][f][r] * gw;
            }
#pragma unroll
            for (int off = 1; off < 16; off <<= 1)
#pragma unroll
                for (int r = 0; r < 8; ++r) p[r] += __shfl_xor(p[r], off, 32);
            if (lm == 0) {
                float gb = gateB[0];
#pragma unroll
                for (int r = 0; r < 8; ++r) {
                    int grow = rowBase + 16 * t + 8 * hb + r;
                    if (grow < R) {
                        float lg = p[r] + gb;
                        logitsOut[grow] = lg;
                        atomicMax(&segMax[colIdx[grow]], fkey(lg));
                    }
                }
            }
        }
    }

    // LDS transpose of D fragments -> coalesced row-major global stores
#pragma unroll
    for (int t = 0; t < 2; ++t)
#pragma unroll
        for (int f = 0; f < 4; ++f)
#pragma unroll
            for (int r = 0; r < 8; ++r)
                lds[(wave * 32 + 16 * t + 8 * hb + r) * 64 + lm + 16 * f] = c[t][f][r];
    __syncthreads();

    if (STATS && threadIdx.x < 64) {
        atomicAdd(&statsSum[threadIdx.x], sSum[threadIdx.x]);
        atomicAdd(&statsSq[threadIdx.x],  sSq[threadIdx.x]);
    }

    for (int i = threadIdx.x; i < 256 * 16; i += 256) {
        int grow = blockIdx.x * 256 + (i >> 4);
        if (grow < R) {
            float4 v = *(const float4*)&lds[i * 4];
            if (OUT_F32)
                ((float4*)outF)[(size_t)grow * 16 + (i & 15)] = v;
            if (OUT_BF16) {
                v4bf o = {(__bf16)v.x, (__bf16)v.y, (__bf16)v.z, (__bf16)v.w};
                ((v4bf*)outBf)[(size_t)grow * 16 + (i & 15)] = o;
            }
        }
    }
}

// ---------- segment softmax: exp(logit - max) and segment sum ----------
__global__ __launch_bounds__(256) void k_exp(const float* __restrict__ logits,
                                             const int* __restrict__ col,
                                             const unsigned* __restrict__ segMaxK, int E,
                                             float* __restrict__ ew, float* __restrict__ segSum) {
    int e = blockIdx.x * 256 + threadIdx.x;
    if (e < E) {
        int cIdx = col[e];
        float w = __expf(logits[e] - funkey(segMaxK[cIdx]));
        ew[e] = w;
        atomicAdd(&segSum[cIdx], w);
    }
}

// ---------- weighted scatter-add to nodes (64 lanes per edge) ----------
__global__ __launch_bounds__(256) void k_scatter(const float* __restrict__ h3,
                                                 const float* __restrict__ ew,
                                                 const float* __restrict__ segSum,
                                                 const int* __restrict__ col, int E,
                                                 float* __restrict__ agg) {
    long long t = (long long)blockIdx.x * 256 + threadIdx.x;
    if (t < (long long)E * 64) {
        int e = (int)(t >> 6), j = (int)(t & 63);
        int c = col[e];
        float coef = ew[e] / segSum[c];
        atomicAdd(&agg[(size_t)c * 64 + j], coef * h3[t]);
    }
}

// ---------- finalize agg (scatter-mean) + BN stats for node MLP layer0 ----------
__global__ __launch_bounds__(64) void k_nodeprep(const float* __restrict__ x,
                                                 const float* __restrict__ agg,
                                                 const int* __restrict__ cnt, int N,
                                                 float* __restrict__ sum, float* __restrict__ sq,
                                                 __bf16* __restrict__ aggb) {
    int j = threadIdx.x;
    float sx = 0.f, qx = 0.f, sa = 0.f, qa = 0.f;
    for (int n = blockIdx.x; n < N; n += gridDim.x) {
        float xv = x[(size_t)n * 64 + j];
        float cdiv = fmaxf((float)cnt[n], 1.f);
        float av = agg[(size_t)n * 64 + j] / cdiv;
        aggb[(size_t)n * 64 + j] = (__bf16)av;
        sx += xv; qx += xv * xv; sa += av; qa += av * av;
    }
    atomicAdd(&sum[j], sx);       atomicAdd(&sq[j], qx);
    atomicAdd(&sum[64 + j], sa);  atomicAdd(&sq[64 + j], qa);
}

// =====================================================================
extern "C" void kernel_launch(void* const* d_in, const int* in_sizes, int n_in,
                              void* d_out, int out_size, void* d_ws, size_t ws_size,
                              hipStream_t stream) {
    const int N = N_NODES, E = N_EDGES;
    const float* x  = (const float*)d_in[0];
    const int*   ei = (const int*)d_in[1];
    const float* ea = (const float*)d_in[2];
    // d_in[3]=u, d_in[4]=batch (unused by reference)
    const int* row = ei;
    const int* col = ei + E;
    // mlp1: g0 be0 w0 b0 g1 be1 w1 b1 g2 be2 w2 b2 -> d_in[5..16]
    // mlp2: same                                  -> d_in[17..28]
    const float* gateW = (const float*)d_in[29];
    const float* gateB = (const float*)d_in[30];
    float* outPtr = (float*)d_out;

    // ---- carve workspace ----
    char* base = (char*)d_ws;
    size_t off = 0;
    auto carve = [&](size_t bytes) -> void* {
        off = (off + 255) & ~(size_t)255;
        void* p = base + off;
        off += bytes;
        return p;
    };
    __bf16* xB   = (__bf16*)carve((size_t)N * 64 * 2);
    __bf16* eaB  = (__bf16*)carve((size_t)E * 64 * 2);   // dead after pass E1
    __bf16* h1B  = (__bf16*)carve((size_t)E * 64 * 2);   // dead after pass E2
    __bf16* h2B  = (__bf16*)carve((size_t)E * 64 * 2);
    float*  h3F  = (float*)eaB;                          // alias: eaB+h1B == E*64*4 bytes
    float*  logi = (float*)carve((size_t)E * 4);
    float*  ew   = (float*)carve((size_t)E * 4);
    int*    degR = (int*)carve((size_t)N * 4);
    int*    cnt  = (int*)carve((size_t)N * 4);
    unsigned* segMax = (unsigned*)carve((size_t)N * 4);
    float*  segSum = (float*)carve((size_t)N * 4);
    float*  agg  = (float*)carve((size_t)N * 64 * 4);
    __bf16* aggB = (__bf16*)carve((size_t)N * 64 * 2);
    __bf16* hn1B = (__bf16*)carve((size_t)N * 64 * 2);
    __bf16* hn2B = (__bf16*)carve((size_t)N * 64 * 2);
    float*  ST   = (float*)carve(1024 * 4);              // packed BN stat accumulators
    __bf16* WP[6]; float* BP[6];
    for (int i = 0; i < 6; ++i) {
        WP[i] = (__bf16*)carve((size_t)64 * 128 * 2);
        BP[i] = (float*)carve(64 * 4);
    }
    (void)ws_size; (void)in_sizes; (void)n_in; (void)out_size;

    // ---- zero accumulators ----
    hipMemsetAsync(ST, 0, 1024 * 4, stream);
    hipMemsetAsync(degR, 0, (size_t)N * 4, stream);
    hipMemsetAsync(cnt, 0, (size_t)N * 4, stream);
    hipMemsetAsync(segMax, 0, (size_t)N * 4, stream);    // 0 < fkey(any finite) used
    hipMemsetAsync(segSum, 0, (size_t)N * 4, stream);
    hipMemsetAsync(agg, 0, (size_t)N * 64 * 4, stream);

    const float invE = 1.0f / (float)E, invN = 1.0f / (float)N;
    const int gridE = (E + 255) / 256, gridN = (N + 255) / 256;

    // ---- BN0 stats + bf16 conversion ----
    k_deg<<<(E + 255) / 256, 256, 0, stream>>>(ei, E, degR, cnt);
    k_statsx<<<256, 64, 0, stream>>>(x, degR, N, ST + 0, ST + 128, xB);
    k_statse<<<512, 64, 0, stream>>>(ea, E, ST + 64, ST + 192, eaB);

    // ---- edge MLP: fold BN -> WMMA GEMM pass, 3 layers ----
    k_fold<<<1, 128, 0, stream>>>(ST + 0, ST + 128, invE, (const float*)d_in[5],
                                  (const float*)d_in[6], (const float*)d_in[7],
                                  (const float*)d_in[8], 128, WP[0], BP[0]);
    gemm_tile<128, true, true, false, true, false><<<gridE, 256, 0, stream>>>(
        xB, eaB, row, WP[0], BP[0], E, h1B, nullptr,
        ST + 256, ST + 320, nullptr, nullptr, nullptr, nullptr, nullptr);

    k_fold<<<1, 128, 0, stream>>>(ST + 256, ST + 320, invE, (const float*)d_in[9],
                                  (const float*)d_in[10], (const float*)d_in[11],
                                  (const float*)d_in[12], 64, WP[1], BP[1]);
    gemm_tile<64, true, true, false, true, false><<<gridE, 256, 0, stream>>>(
        h1B, nullptr, nullptr, WP[1], BP[1], E, h2B, nullptr,
        ST + 384, ST + 448, nullptr, nullptr, nullptr, nullptr, nullptr);

    k_fold<<<1, 128, 0, stream>>>(ST + 384, ST + 448, invE, (const float*)d_in[13],
                                  (const float*)d_in[14], (const float*)d_in[15],
                                  (const float*)d_in[16], 64, WP[2], BP[2]);
    gemm_tile<64, false, false, true, false, true><<<gridE, 256, 0, stream>>>(
        h2B, nullptr, nullptr, WP[2], BP[2], E, nullptr, h3F,
        nullptr, nullptr, gateW, gateB, logi, segMax, col);

    // ---- segment softmax + scatter-mean ----
    k_exp<<<(E + 255) / 256, 256, 0, stream>>>(logi, col, segMax, E, ew, segSum);
    k_scatter<<<(int)(((long long)E * 64 + 255) / 256), 256, 0, stream>>>(
        h3F, ew, segSum, col, E, agg);
    k_nodeprep<<<256, 64, 0, stream>>>(x, agg, cnt, N, ST + 512, ST + 640, aggB);

    // ---- node MLP: 3 layers ----
    k_fold<<<1, 128, 0, stream>>>(ST + 512, ST + 640, invN, (const float*)d_in[17],
                                  (const float*)d_in[18], (const float*)d_in[19],
                                  (const float*)d_in[20], 128, WP[3], BP[3]);
    gemm_tile<128, true, true, false, true, false><<<gridN, 256, 0, stream>>>(
        xB, aggB, nullptr, WP[3], BP[3], N, hn1B, nullptr,
        ST + 768, ST + 832, nullptr, nullptr, nullptr, nullptr, nullptr);

    k_fold<<<1, 128, 0, stream>>>(ST + 768, ST + 832, invN, (const float*)d_in[21],
                                  (const float*)d_in[22], (const float*)d_in[23],
                                  (const float*)d_in[24], 64, WP[4], BP[4]);
    gemm_tile<64, true, true, false, true, false><<<gridN, 256, 0, stream>>>(
        hn1B, nullptr, nullptr, WP[4], BP[4], N, hn2B, nullptr,
        ST + 896, ST + 960, nullptr, nullptr, nullptr, nullptr, nullptr);

    k_fold<<<1, 128, 0, stream>>>(ST + 896, ST + 960, invN, (const float*)d_in[25],
                                  (const float*)d_in[26], (const float*)d_in[27],
                                  (const float*)d_in[28], 64, WP[5], BP[5]);
    gemm_tile<64, false, false, true, false, false><<<gridN, 256, 0, stream>>>(
        hn2B, nullptr, nullptr, WP[5], BP[5], N, nullptr, outPtr,
        nullptr, nullptr, nullptr, nullptr, nullptr, nullptr, nullptr);
}
